// sn_cc_layerwise_44006234914837
// MI455X (gfx1250) — matrix operand order repaired and checked by reading, not compiled, not run on hardware
//
#include <hip/hip_runtime.h>

// ---------------------------------------------------------------------------
// Types / helpers
// ---------------------------------------------------------------------------
typedef __bf16 bf16;
typedef __attribute__((ext_vector_type(8)))  __bf16 v8bf;
typedef __attribute__((ext_vector_type(16))) __bf16 v16bf;
typedef __attribute__((ext_vector_type(8)))  float  v8f;

#define BATCH 32

__device__ __forceinline__ float bf2f(bf16 x) {
  unsigned short u = __builtin_bit_cast(unsigned short, x);
  unsigned int v = ((unsigned int)u) << 16;
  return __builtin_bit_cast(float, v);
}
__device__ __forceinline__ bf16 f2bf(float f) {
  unsigned int u = __builtin_bit_cast(unsigned int, f);
  u += 0x7fffu + ((u >> 16) & 1u);          // round-to-nearest-even
  unsigned short h = (unsigned short)(u >> 16);
  return __builtin_bit_cast(bf16, h);
}
__device__ __forceinline__ v16bf pack16(v8bf lo, v8bf hi) {
  v16bf r;
#pragma unroll
  for (int i = 0; i < 8; ++i) { r[i] = lo[i]; r[8 + i] = hi[i]; }
  return r;
}
__device__ __forceinline__ v16bf zero16() {
  v16bf r;
#pragma unroll
  for (int i = 0; i < 16; ++i) r[i] = __builtin_bit_cast(bf16, (unsigned short)0);
  return r;
}

// ---------------------------------------------------------------------------
// Prep kernels
// ---------------------------------------------------------------------------
__global__ void bn_prep_kernel(const float* __restrict__ g, const float* __restrict__ b,
                               const float* __restrict__ m, const float* __restrict__ v,
                               float* __restrict__ scale, float* __restrict__ shift, int C) {
  int i = blockIdx.x * blockDim.x + threadIdx.x;
  if (i < C) {
    float inv = g[i] * rsqrtf(v[i] + 1e-5f);
    scale[i] = inv;
    shift[i] = b[i] - m[i] * inv;
  }
}

// OIHW f32 -> [RS][O][I] bf16 (per-tap weight slices, Cin contiguous)
__global__ void wprep_kernel(const float* __restrict__ src, bf16* __restrict__ dst,
                             int O, int I, int RS) {
  int idx = blockIdx.x * blockDim.x + threadIdx.x;
  int total = O * I * RS;
  if (idx >= total) return;
  int i  = idx % I;
  int t  = idx / I;
  int o  = t % O;
  int rs = t / O;
  dst[idx] = f2bf(src[((size_t)o * I + i) * RS + rs]);
}

// conv1 weights: sum over the 3 (identical-data) input channels -> [64][49] f32
__global__ void w1sum_kernel(const float* __restrict__ w, float* __restrict__ dst) {
  int idx = blockIdx.x * blockDim.x + threadIdx.x;
  if (idx >= 64 * 49) return;
  int o = idx / 49, rs = idx % 49;
  dst[idx] = w[(o * 3 + 0) * 49 + rs] + w[(o * 3 + 1) * 49 + rs] + w[(o * 3 + 2) * 49 + rs];
}

// ---------------------------------------------------------------------------
// conv1: 7x7 s2 p3, single input channel, fused BN+ReLU, out NHWC bf16
// grid = B*150*150, block = 64 (cout)
// ---------------------------------------------------------------------------
__global__ void conv1_direct_kernel(const float* __restrict__ img, const float* __restrict__ w1s,
                                    const float* __restrict__ scale, const float* __restrict__ shift,
                                    bf16* __restrict__ out) {
  int cout = threadIdx.x;
  int tmp = blockIdx.x;
  int wo = tmp % 150; tmp /= 150;
  int ho = tmp % 150;
  int b  = tmp / 150;
  float acc = 0.f;
  const float* wrow = w1s + cout * 49;
  const float* ib = img + (size_t)b * 299 * 299;
#pragma unroll
  for (int r = 0; r < 7; ++r) {
    int y = ho * 2 + r - 3;
    if (y < 0 || y >= 299) continue;
#pragma unroll
    for (int s = 0; s < 7; ++s) {
      int x = wo * 2 + s - 3;
      if (x < 0 || x >= 299) continue;
      acc += ib[y * 299 + x] * wrow[r * 7 + s];
    }
  }
  float v = fmaxf(acc * scale[cout] + shift[cout], 0.f);
  out[(size_t)blockIdx.x * 64 + cout] = f2bf(v);
}

// maxpool 3x3 s2 p1 NHWC bf16; grid = B*Ho*Ho, block = C
__global__ void maxpool_kernel(const bf16* __restrict__ in, bf16* __restrict__ out,
                               int Hi, int Ho, int C) {
  int c = threadIdx.x;
  int tmp = blockIdx.x;
  int wo = tmp % Ho; tmp /= Ho;
  int ho = tmp % Ho;
  int b  = tmp / Ho;
  float m = -3.0e38f;
#pragma unroll
  for (int dy = 0; dy < 3; ++dy) {
    int y = ho * 2 + dy - 1;
    if (y < 0 || y >= Hi) continue;
#pragma unroll
    for (int dx = 0; dx < 3; ++dx) {
      int x = wo * 2 + dx - 1;
      if (x < 0 || x >= Hi) continue;
      m = fmaxf(m, bf2f(in[(((size_t)b * Hi + y) * Hi + x) * C + c]));
    }
  }
  out[(size_t)blockIdx.x * C + c] = f2bf(m);
}

// ---------------------------------------------------------------------------
// WMMA implicit-GEMM conv v2 (bf16 NHWC in, [RS][Co][Ci] bf16 weights)
//   Each wave: M = 16 output positions (its own tile) x N = 64 couts (4 WMMAs/K-step)
//   Weights for the block's 64-cout slab staged in LDS per tap, shared by 4 waves.
//   block = 128 (4 waves = 4 position tiles), grid = (ceil(nPosTiles/4), Co/64)
//   mode: 0 = bn only, 1 = bn+relu, 2 = bn+residual+relu
// ---------------------------------------------------------------------------
__global__ void __launch_bounds__(128)
conv_wmma_kernel(const bf16* __restrict__ in, const bf16* __restrict__ wt,
                 const float* __restrict__ scale, const float* __restrict__ shift,
                 const bf16* __restrict__ res, bf16* __restrict__ out,
                 int Hi, int Wi, int Ci, int Ho, int Wo, int Co,
                 int R, int S, int stride, int pad, int nWoTiles, int nPosTiles, int mode) {
  extern __shared__ bf16 smem[];          // [64][Ci + 8] (padded rows, bank spread)
  const int pitch = Ci + 8;

  int lane = threadIdx.x & 31;
  int wave = threadIdx.x >> 5;
  int coutSlab = blockIdx.y * 64;

  int posTile = blockIdx.x * 4 + wave;    // per-wave position tile
  bool active = (posTile < nPosTiles);
  int pt = active ? posTile : 0;
  int woTile = pt % nWoTiles;
  int t2 = pt / nWoTiles;
  int ho = t2 % Ho;
  int b  = t2 / Ho;
  int woBase = woTile * 16;

  // A fragment (16x32 bf16): lane -> M = lane&15; K halves split per ISA layout
  int mA    = lane & 15;
  int koffA = (lane & 16) ? 8 : 0;        // lanes 16-31 hold K 8-15 / 24-31
  int wo    = woBase + mA;
  bool colOk = (wo < Wo);

  // B fragment (32x16 bf16): lane -> N = lane&15; lanes 16-31 hold K 8-15 / 24-31
  int nIdx  = lane & 15;
  int koffB = (lane & 16) ? 8 : 0;

  v8f acc[4];
#pragma unroll
  for (int t = 0; t < 4; ++t)
#pragma unroll
    for (int i = 0; i < 8; ++i) acc[t][i] = 0.f;

  int kc32 = Ci >> 5;
  int RS = R * S;
  int chunksPerRow = Ci >> 3;             // 8-half chunks per cout row
  int totalChunks  = chunksPerRow * 64;

  for (int rs = 0; rs < RS; ++rs) {
    // ---- cooperative stage of this tap's 64xCi weight slab into LDS ----
    for (int cidx = threadIdx.x; cidx < totalChunks; cidx += 128) {
      int row = cidx / chunksPerRow;
      int col = (cidx - row * chunksPerRow) * 8;
      v8bf wv = *(const v8bf*)(wt + ((size_t)rs * Co + coutSlab + row) * Ci + col);
      *(v8bf*)(&smem[row * pitch + col]) = wv;
    }
    __syncthreads();

    int r = rs / S, s = rs - r * S;
    int hi = ho * stride + r - pad;       // wave-uniform
    if (active && hi >= 0 && hi < Hi) {
      int wi = wo * stride + s - pad;     // per-lane
      bool aOk = colOk && (wi >= 0) && (wi < Wi);
      const bf16* aBase = in + (((size_t)b * Hi + hi) * Wi + wi) * Ci + koffA;
      for (int kc = 0; kc < kc32; ++kc) {
        v16bf afrag;
        if (aOk) {
          v8bf a0 = *(const v8bf*)(aBase + kc * 32);
          v8bf a1 = *(const v8bf*)(aBase + kc * 32 + 16);
          afrag = pack16(a0, a1);
        } else {
          afrag = zero16();
        }
#pragma unroll
        for (int t = 0; t < 4; ++t) {
          const bf16* bp = &smem[(t * 16 + nIdx) * pitch + kc * 32 + koffB];
          v8bf b0 = *(const v8bf*)(bp);
          v8bf b1 = *(const v8bf*)(bp + 16);
          v16bf bfrag = pack16(b0, b1);
          acc[t] = __builtin_amdgcn_wmma_f32_16x16x32_bf16(
              false, afrag, false, bfrag, (short)0, acc[t], false, false);
        }
      }
    }
    __syncthreads();                      // LDS reused next tap
  }

  if (!active) return;

  // Epilogue: C/D layout -> VGPR rr: M = rr + (lane>=16 ? 8 : 0), N = lane&15
  int mBase = (lane & 16) ? 8 : 0;
#pragma unroll
  for (int t = 0; t < 4; ++t) {
    int cout = coutSlab + t * 16 + nIdx;
    float sc = scale[cout], sh = shift[cout];
#pragma unroll
    for (int rr = 0; rr < 8; ++rr) {
      int woO = woBase + rr + mBase;
      if (woO >= Wo) continue;
      float v = acc[t][rr] * sc + sh;
      size_t oidx = (((size_t)b * Ho + ho) * Wo + woO) * Co + cout;
      if (mode == 2) v += bf2f(res[oidx]);
      if (mode != 0) v = fmaxf(v, 0.f);
      out[oidx] = f2bf(v);
    }
  }
}

// ---------------------------------------------------------------------------
// Per-sample cross-correlation, summed over channels.
// fm1/fm2: NHWC bf16 [B,H,H,C]; out f32 [B,P,P]; grid = B*P*P, block = 256
// ---------------------------------------------------------------------------
__global__ void xcross_kernel(const bf16* __restrict__ fm1, const bf16* __restrict__ fm2,
                              float* __restrict__ out, int H, int C, int P, int pad) {
  __shared__ float red[256];
  int tmp = blockIdx.x;
  int q = tmp % P; tmp /= P;
  int p = tmp % P;
  int b = tmp / P;
  int u0 = max(0, pad - p), u1 = min(H, H + pad - p);
  int v0 = max(0, pad - q), v1 = min(H, H + pad - q);
  int uc = u1 - u0, vc = v1 - v0;
  int c8 = C >> 3;
  int total = uc * vc * c8;
  const bf16* f1b = fm1 + (size_t)b * H * H * C;
  const bf16* f2b = fm2 + (size_t)b * H * H * C;
  float acc = 0.f;
  for (int i = threadIdx.x; i < total; i += 256) {
    int ci = i % c8;
    int t2 = i / c8;
    int vi = t2 % vc;
    int ui = t2 / vc;
    int u = u0 + ui, v = v0 + vi;
    int y = p + u - pad, x = q + v - pad;
    v8bf a = *(const v8bf*)(f1b + ((size_t)y * H + x) * C + ci * 8);
    v8bf w = *(const v8bf*)(f2b + ((size_t)u * H + v) * C + ci * 8);
#pragma unroll
    for (int k = 0; k < 8; ++k) acc += bf2f(a[k]) * bf2f(w[k]);
  }
  red[threadIdx.x] = acc;
  __syncthreads();
  for (int sft = 128; sft > 0; sft >>= 1) {
    if (threadIdx.x < sft) red[threadIdx.x] += red[threadIdx.x + sft];
    __syncthreads();
  }
  if (threadIdx.x == 0) out[blockIdx.x] = red[0];
}

// adaptive avg pool 39->19 (both dims) + concat + relu -> feat[B][722]
// grid = B, block = 384
__global__ void poolcat_kernel(const float* __restrict__ sm38, const float* __restrict__ sm19,
                               float* __restrict__ feat) {
  int b = blockIdx.x;
  int t = threadIdx.x;
  if (t < 361) {
    int i = t / 19, j = t % 19;
    int s0 = (i * 39) / 19, e0 = ((i + 1) * 39 + 18) / 19;
    int s1 = (j * 39) / 19, e1 = ((j + 1) * 39 + 18) / 19;
    float sum = 0.f;
    for (int a = s0; a < e0; ++a)
      for (int c = s1; c < e1; ++c)
        sum += sm38[(size_t)b * 1521 + a * 39 + c];
    sum /= (float)((e0 - s0) * (e1 - s1));
    feat[b * 722 + t] = fmaxf(sum, 0.f);
    feat[b * 722 + 361 + t] = fmaxf(sm19[b * 361 + t], 0.f);
  }
}

// fc1 (722->256) + relu; grid = B, block = 256
__global__ void fc1_kernel(const float* __restrict__ feat, const float* __restrict__ w,
                           const float* __restrict__ bias, float* __restrict__ out) {
  int b = blockIdx.x;
  int o = threadIdx.x;
  float acc = bias[o];
  const float* f = feat + b * 722;
  const float* wr = w + o * 722;
  for (int i = 0; i < 722; ++i) acc += f[i] * wr[i];
  out[b * 256 + o] = fmaxf(acc, 0.f);
}

// fcf (256->2); grid = B, block = 32
__global__ void fcf_kernel(const float* __restrict__ h, const float* __restrict__ w,
                           const float* __restrict__ bias, float* __restrict__ out) {
  int b = blockIdx.x;
  int o = threadIdx.x;
  if (o < 2) {
    float acc = bias[o];
    for (int i = 0; i < 256; ++i) acc += h[b * 256 + i] * w[o * 256 + i];
    out[b * 2 + o] = acc;
  }
}

// ---------------------------------------------------------------------------
// Host orchestration
// ---------------------------------------------------------------------------
struct BNP { float* s; float* h; };

extern "C" void kernel_launch(void* const* d_in, const int* in_sizes, int n_in,
                              void* d_out, int out_size, void* d_ws, size_t ws_size,
                              hipStream_t stream) {
  (void)in_sizes; (void)out_size; (void)ws_size;
  if (n_in < 81) return;

  // jax pytree leaf order (dict keys sorted at every level)
  enum {
    IN_IMG1 = 0, IN_IMG2 = 1,
    P_BN1 = 2,                                    // b,g,m,v
    P_CONV1_W = 6,
    P_FC1_B = 7, P_FC1_W = 8, P_FCF_B = 9, P_FCF_W = 10,
    L1B0_BN1 = 11, L1B0_BN2 = 15, L1B0_C1 = 19, L1B0_C2 = 20,
    L1B1_BN1 = 21, L1B1_BN2 = 25, L1B1_C1 = 29, L1B1_C2 = 30,
    L2B0_BN1 = 31, L2B0_BN2 = 35, L2B0_C1 = 39, L2B0_C2 = 40, L2B0_DBN = 41, L2B0_DW = 45,
    L2B1_BN1 = 46, L2B1_BN2 = 50, L2B1_C1 = 54, L2B1_C2 = 55,
    L3B0_BN1 = 56, L3B0_BN2 = 60, L3B0_C1 = 64, L3B0_C2 = 65, L3B0_DBN = 66, L3B0_DW = 70,
    L3B1_BN1 = 71, L3B1_BN2 = 75, L3B1_C1 = 79, L3B1_C2 = 80
  };

  size_t cur = 0;
  auto alloc = [&](size_t bytes) -> void* {
    void* p = (char*)d_ws + cur;
    cur = (cur + bytes + 255) & ~(size_t)255;
    return p;
  };

  auto prepBN = [&](int base, int C) -> BNP {
    float* sc = (float*)alloc((size_t)C * 4);
    float* sh = (float*)alloc((size_t)C * 4);
    bn_prep_kernel<<<(C + 63) / 64, 64, 0, stream>>>(
        (const float*)d_in[base + 1], (const float*)d_in[base + 0],
        (const float*)d_in[base + 2], (const float*)d_in[base + 3], sc, sh, C);
    return {sc, sh};
  };
  auto prepW = [&](int idx, int O, int I, int RS) -> bf16* {
    bf16* dst = (bf16*)alloc((size_t)O * I * RS * 2);
    int total = O * I * RS;
    wprep_kernel<<<(total + 255) / 256, 256, 0, stream>>>((const float*)d_in[idx], dst, O, I, RS);
    return dst;
  };

  // ---- parameter prep ----
  BNP bnTop = prepBN(P_BN1, 64);
  BNP bn10a = prepBN(L1B0_BN1, 64),  bn10b = prepBN(L1B0_BN2, 64);
  BNP bn11a = prepBN(L1B1_BN1, 64),  bn11b = prepBN(L1B1_BN2, 64);
  BNP bn20a = prepBN(L2B0_BN1, 128), bn20b = prepBN(L2B0_BN2, 128), bn20d = prepBN(L2B0_DBN, 128);
  BNP bn21a = prepBN(L2B1_BN1, 128), bn21b = prepBN(L2B1_BN2, 128);
  BNP bn30a = prepBN(L3B0_BN1, 256), bn30b = prepBN(L3B0_BN2, 256), bn30d = prepBN(L3B0_DBN, 256);
  BNP bn31a = prepBN(L3B1_BN1, 256), bn31b = prepBN(L3B1_BN2, 256);

  float* w1s = (float*)alloc(64 * 49 * 4);
  w1sum_kernel<<<(64 * 49 + 255) / 256, 256, 0, stream>>>((const float*)d_in[P_CONV1_W], w1s);

  bf16* w10c1 = prepW(L1B0_C1, 64, 64, 9);
  bf16* w10c2 = prepW(L1B0_C2, 64, 64, 9);
  bf16* w11c1 = prepW(L1B1_C1, 64, 64, 9);
  bf16* w11c2 = prepW(L1B1_C2, 64, 64, 9);
  bf16* w20c1 = prepW(L2B0_C1, 128, 64, 9);
  bf16* w20c2 = prepW(L2B0_C2, 128, 128, 9);
  bf16* w20d  = prepW(L2B0_DW, 128, 64, 1);
  bf16* w21c1 = prepW(L2B1_C1, 128, 128, 9);
  bf16* w21c2 = prepW(L2B1_C2, 128, 128, 9);
  bf16* w30c1 = prepW(L3B0_C1, 256, 128, 9);
  bf16* w30c2 = prepW(L3B0_C2, 256, 256, 9);
  bf16* w30d  = prepW(L3B0_DW, 256, 128, 1);
  bf16* w31c1 = prepW(L3B1_C1, 256, 256, 9);
  bf16* w31c2 = prepW(L3B1_C2, 256, 256, 9);

  // ---- activation buffers ----
  bf16* A0 = (bf16*)alloc((size_t)BATCH * 150 * 150 * 64 * 2);   // conv1 out
  bf16* G0 = (bf16*)alloc((size_t)BATCH * 75 * 75 * 64 * 2);
  bf16* G1 = (bf16*)alloc((size_t)BATCH * 75 * 75 * 64 * 2);
  bf16* G2 = (bf16*)alloc((size_t)BATCH * 75 * 75 * 64 * 2);
  bf16* G3 = (bf16*)alloc((size_t)BATCH * 75 * 75 * 64 * 2);
  bf16* FM38A = (bf16*)alloc((size_t)BATCH * 38 * 38 * 128 * 2);
  bf16* FM38B = (bf16*)alloc((size_t)BATCH * 38 * 38 * 128 * 2);
  bf16* FM19A = (bf16*)alloc((size_t)BATCH * 19 * 19 * 256 * 2);
  bf16* FM19B = (bf16*)alloc((size_t)BATCH * 19 * 19 * 256 * 2);
  float* SM38 = (float*)alloc((size_t)BATCH * 39 * 39 * 4);
  float* SM19 = (float*)alloc((size_t)BATCH * 19 * 19 * 4);
  float* FEAT = (float*)alloc((size_t)BATCH * 722 * 4);
  float* FC1O = (float*)alloc((size_t)BATCH * 256 * 4);

  auto convW = [&](const bf16* in, const bf16* wt, BNP bn, const bf16* res, bf16* out,
                   int Hi, int Ci, int Ho, int Co, int R, int stride, int pad, int mode) {
    int nWoT = (Ho + 15) / 16;                 // Wo == Ho (square maps)
    int nPosTiles = BATCH * Ho * nWoT;
    dim3 grid((unsigned)((nPosTiles + 3) / 4), (unsigned)(Co / 64));
    size_t shbytes = (size_t)(Ci + 8) * 64 * 2;  // padded 64-row weight slab
    conv_wmma_kernel<<<grid, 128, shbytes, stream>>>(in, wt, bn.s, bn.h, res, out,
        Hi, Hi, Ci, Ho, Ho, Co, R, R, stride, pad, nWoT, nPosTiles, mode);
  };
  auto runBlock = [&](const bf16* x, bf16* t1, bf16* out, int H, int C,
                      bf16* wA, BNP bA, bf16* wB, BNP bB) {
    convW(x,  wA, bA, nullptr, t1,  H, C, H, C, 3, 1, 1, 1);
    convW(t1, wB, bB, x,       out, H, C, H, C, 3, 1, 1, 2);
  };
  auto runBlockDown = [&](const bf16* x, bf16* t1, bf16* resb, bf16* out,
                          int Hi, int Ci, int Ho, int Co,
                          bf16* wA, BNP bA, bf16* wB, BNP bB, bf16* wD, BNP bD) {
    convW(x,  wA, bA, nullptr, t1,   Hi, Ci, Ho, Co, 3, 2, 1, 1);
    convW(x,  wD, bD, nullptr, resb, Hi, Ci, Ho, Co, 1, 2, 0, 0);
    convW(t1, wB, bB, resb,    out,  Ho, Co, Ho, Co, 3, 1, 1, 2);
  };

  auto backbone = [&](const float* img, bf16* fm38, bf16* fm19) {
    conv1_direct_kernel<<<BATCH * 150 * 150, 64, 0, stream>>>(img, w1s, bnTop.s, bnTop.h, A0);
    maxpool_kernel<<<BATCH * 75 * 75, 64, 0, stream>>>(A0, G0, 150, 75, 64);
    runBlock(G0, G1, G2, 75, 64, w10c1, bn10a, w10c2, bn10b);        // layer1[0]
    runBlock(G2, G0, G1, 75, 64, w11c1, bn11a, w11c2, bn11b);        // layer1[1]
    runBlockDown(G1, G2, G0, G3, 75, 64, 38, 128,
                 w20c1, bn20a, w20c2, bn20b, w20d, bn20d);           // layer2[0]
    runBlock(G3, G1, fm38, 38, 128, w21c1, bn21a, w21c2, bn21b);     // layer2[1] -> fm38
    runBlockDown(fm38, G0, G1, G2, 38, 128, 19, 256,
                 w30c1, bn30a, w30c2, bn30b, w30d, bn30d);           // layer3[0]
    runBlock(G2, G0, fm19, 19, 256, w31c1, bn31a, w31c2, bn31b);     // layer3[1] -> fm19
  };

  backbone((const float*)d_in[IN_IMG1], FM38A, FM19A);
  backbone((const float*)d_in[IN_IMG2], FM38B, FM19B);

  xcross_kernel<<<BATCH * 39 * 39, 256, 0, stream>>>(FM38A, FM38B, SM38, 38, 128, 39, 19);
  xcross_kernel<<<BATCH * 19 * 19, 256, 0, stream>>>(FM19A, FM19B, SM19, 19, 256, 19, 9);

  poolcat_kernel<<<BATCH, 384, 0, stream>>>(SM38, SM19, FEAT);
  fc1_kernel<<<BATCH, 256, 0, stream>>>(FEAT, (const float*)d_in[P_FC1_W],
                                        (const float*)d_in[P_FC1_B], FC1O);
  fcf_kernel<<<BATCH, 32, 0, stream>>>(FC1O, (const float*)d_in[P_FCF_W],
                                       (const float*)d_in[P_FCF_B], (float*)d_out);
}
